// STMGT_35278861369728
// MI455X (gfx1250) — matrix operand order repaired and checked by reading, not compiled, not run on hardware
//
#include <hip/hip_runtime.h>
#include <cmath>

// ---------------- problem constants ----------------
#define BB      4
#define NN      300
#define TT      24
#define TWW     24
#define HH      64
#define NHEADS  4
#define NBLK    3
#define OUTC    36        // PRED*K = 12*3
#define NEDGE   3900      // E + N self loops
#define NGRAPH  (BB*TT)   // 96
#define NROWS   (BB*NN*TT)// 28800
#define LN_EPS  1e-5f
#define PI_F    3.14159265358979323846f
#define NTILE   4         // N-tiles per wave in wmma_linear (all N % 64 == 0)

typedef __attribute__((ext_vector_type(16))) _Float16 v16h;
typedef __attribute__((ext_vector_type(8)))  float    v8f;
typedef __attribute__((ext_vector_type(4)))  float    v4f;

// ---------------- helpers ----------------
__device__ inline float gelu_exact(float x) {
    return 0.5f * x * (1.0f + erff(x * 0.70710678118654752f));
}

__device__ inline float blocksum64(float v, float* red) {
    int tid = threadIdx.x;
    red[tid] = v; __syncthreads();
    #pragma unroll
    for (int s = 32; s > 0; s >>= 1) {
        if (tid < s) red[tid] += red[tid + s];
        __syncthreads();
    }
    float r = red[0]; __syncthreads();
    return r;
}

__device__ inline float ln_norm64(float v, const float* g, const float* b,
                                  int d, float* red) {
    float mean = blocksum64(v, red) * (1.0f / 64.0f);
    float c = v - mean;
    float var = blocksum64(c * c, red) * (1.0f / 64.0f);
    return c * rsqrtf(var + LN_EPS) * g[d] + b[d];
}

__device__ inline void atomicMaxF(float* addr, float val) {
    int* ia = (int*)addr;
    int old = __float_as_int(*addr);
    while (__int_as_float(old) < val) {
        int assumed = old;
        old = atomicCAS(ia, assumed, __float_as_int(val));
        if (old == assumed) break;
    }
}

// ---------------- weight pre-conversion f32 -> f16 ----------------
__global__ void f32_to_f16(const float* __restrict__ src, _Float16* __restrict__ dst, int n) {
    int i = blockIdx.x * blockDim.x + threadIdx.x;
    if (i < n) dst[i] = (_Float16)src[i];
}

// ---------------- WMMA linear: out[M,N] = act(X[M,K] @ W[N,K(ldw)]^T + bias) ----------------
// W pre-converted to f16. One wave32 per (16-row M tile) x (4 x 16-col N tiles).
// A fragment loaded once per K-step (f32 -> cvt), 4 B fragments loaded directly
// as packed f16 (v16h = 2 x b128) into distinct registers, 4 unrolled WMMAs.
// Requires M%16==0, N%64==0, K%32==0 (all call sites satisfy this).
__global__ void __launch_bounds__(128) wmma_linear(
    const float* __restrict__ X, const _Float16* __restrict__ Wh,
    const float* __restrict__ bias, float* __restrict__ out,
    int M, int N, int K, int ldw, int accum, int act)
{
    const int lane   = threadIdx.x & 31;
    const int wave   = blockIdx.x * (blockDim.x >> 5) + (threadIdx.x >> 5);
    const int tilesM = M >> 4;
    const int groupsN = N >> 6;              // groups of 4 N-tiles
    if (wave >= tilesM * groupsN) return;
    const int tM = wave / groupsN, tG = wave % groupsN;
    const int lh = lane & 15;
    const int hi = (lane >> 4) & 1;
    const int m  = (tM << 4) + lh;
    const float* xrow = X + (size_t)m * K;
    const int kbA = hi ? 16 : 0;             // A frag: K = kbA + j
    const int kbB = hi ? 16 : 0;             // B frag: K = kbB + j

    const _Float16* wrow[NTILE];
    int ncol[NTILE];
    #pragma unroll
    for (int nt = 0; nt < NTILE; ++nt) {
        ncol[nt] = ((tG * NTILE + nt) << 4) + lh;
        wrow[nt] = Wh + (size_t)ncol[nt] * ldw;
    }

    v8f acc[NTILE] = {};
    for (int k0 = 0; k0 < K; k0 += 32) {
        // ---- issue all loads first (distinct registers) ----
        const v4f* pa0 = (const v4f*)(xrow + k0 + kbA);
        const v4f* pa1 = (const v4f*)(xrow + k0 + 8 + kbA);
        v4f a0 = pa0[0], a1 = pa0[1], a2 = pa1[0], a3 = pa1[1];
        v16h bv[NTILE];
        #pragma unroll
        for (int nt = 0; nt < NTILE; ++nt)
            bv[nt] = *(const v16h*)(wrow[nt] + k0 + kbB);
        // ---- A fragment conversion ----
        v16h av;
        #pragma unroll
        for (int j = 0; j < 4; ++j) {
            av[j]      = (_Float16)a0[j];
            av[4 + j]  = (_Float16)a1[j];
            av[8 + j]  = (_Float16)a2[j];
            av[12 + j] = (_Float16)a3[j];
        }
        // ---- 4 WMMAs, A reused ----
        #pragma unroll
        for (int nt = 0; nt < NTILE; ++nt)
            acc[nt] = __builtin_amdgcn_wmma_f32_16x16x32_f16(
                false, av, false, bv[nt], (short)0, acc[nt], false, false);
    }

    const int rowbase = (tM << 4) + (hi ? 8 : 0);
    #pragma unroll
    for (int nt = 0; nt < NTILE; ++nt) {
        float bb = bias ? bias[ncol[nt]] : 0.0f;
        #pragma unroll
        for (int r = 0; r < 8; ++r) {
            size_t oidx = (size_t)(rowbase + r) * N + ncol[nt];
            float v = acc[nt][r] + bb;
            if (accum) v += out[oidx];
            if (act == 1) v = gelu_exact(v);
            out[oidx] = v;
        }
    }
}

// ---------------- small scalar linear (K or N not WMMA friendly) ----------------
// op: 0=none, 1=clip(exp(x),0.1,10)
__global__ void small_linear(const float* __restrict__ X, const float* __restrict__ W,
                             const float* __restrict__ bias, float* __restrict__ out,
                             int M, int Nout, int K, int op)
{
    int i = blockIdx.x * blockDim.x + threadIdx.x;
    if (i >= M * Nout) return;
    int r = i / Nout, n = i % Nout;
    float acc = bias ? bias[n] : 0.0f;
    for (int k = 0; k < K; ++k) acc += X[(size_t)r * K + k] * W[(size_t)n * K + k];
    if (op == 1) { acc = expf(acc); acc = fminf(fmaxf(acc, 0.1f), 10.0f); }
    out[i] = acc;
}

// ---------------- fill ----------------
__global__ void fillf(float* p, float v, size_t n) {
    size_t i = (size_t)blockIdx.x * blockDim.x + threadIdx.x;
    if (i < n) p[i] = v;
}

// ---------------- temporal embedding: temp[B*T,64] ----------------
__global__ void temp_embed(const float* hour, const int* dow, const int* is_we,
                           const float* dow_tab, const float* we_tab,
                           const float* tproj_W, const float* tproj_b,
                           float* temp)
{
    int bt  = blockIdx.x;
    int tid = threadIdx.x;
    __shared__ float feat[50];
    if (tid == 0) {
        float hr = hour[bt] * (2.0f * PI_F / 24.0f);
        feat[0] = sinf(hr); feat[1] = cosf(hr);
    }
    int dw = dow[bt], we = is_we[bt];
    if (tid >= 2 && tid < 34)  feat[tid] = dow_tab[dw * 32 + (tid - 2)];
    if (tid >= 34 && tid < 50) feat[tid] = we_tab[we * 16 + (tid - 34)];
    __syncthreads();
    float acc = tproj_b[tid];
    for (int k = 0; k < 50; ++k) acc += feat[k] * tproj_W[tid * 50 + k];
    temp[bt * 64 + tid] = acc;
}

// ---------------- encoder: x = traffic*enc_W + enc_b + temp ----------------
__global__ void encode(const float* x_traffic, const float* enc_W, const float* enc_b,
                       const float* temp, float* x)
{
    int r = blockIdx.x;            // (b*NN+n)*TT + t
    int h = threadIdx.x;
    int t = r % TT; int b = (r / TT) / NN;
    x[(size_t)r * 64 + h] = x_traffic[r] * enc_W[h] + enc_b[h] + temp[(b * TT + t) * 64 + h];
}

// ---------------- GATv2 edge logits + segment max ----------------
__global__ void gat_edge_logits(const float* __restrict__ xl, const float* __restrict__ xr,
                                const float* __restrict__ att,
                                const int* __restrict__ src, const int* __restrict__ dst,
                                float* __restrict__ logits, float* __restrict__ segmax)
{
    int i = blockIdx.x * blockDim.x + threadIdx.x;
    if (i >= NGRAPH * NEDGE * NHEADS) return;
    int h = i & 3; int ge = i >> 2;
    int e = ge % NEDGE; int g = ge / NEDGE;
    int b = g / TT, t = g % TT;
    int s = src[e], d = dst[e];
    const float* pl = xl + ((size_t)(b * NN + s) * TT + t) * 256 + h * 64;
    const float* pr = xr + ((size_t)(b * NN + d) * TT + t) * 256 + h * 64;
    const float* pa = att + h * 64;
    float acc = 0.0f;
    for (int k = 0; k < 64; ++k) {
        float ev = pl[k] + pr[k];
        ev = (ev > 0.0f) ? ev : 0.2f * ev;       // leaky_relu 0.2
        acc += ev * pa[k];
    }
    logits[i] = acc;
    atomicMaxF(&segmax[(g * NN + d) * NHEADS + h], acc);
}

// ---------------- GATv2 edge accumulate ----------------
__global__ void gat_edge_accum(const float* __restrict__ xl,
                               const int* __restrict__ src, const int* __restrict__ dst,
                               const float* __restrict__ logits, const float* __restrict__ segmax,
                               float* __restrict__ denom, float* __restrict__ msg)
{
    long long i = (long long)blockIdx.x * blockDim.x + threadIdx.x;
    if (i >= (long long)NGRAPH * NEDGE * NHEADS * 8) return;
    int c = (int)(i & 7); long long rest = i >> 3;
    int h = (int)(rest & 3); long long ge = rest >> 2;
    int e = (int)(ge % NEDGE); int g = (int)(ge / NEDGE);
    int b = g / TT, t = g % TT;
    int s = src[e], d = dst[e];
    float w = expf(logits[((size_t)g * NEDGE + e) * NHEADS + h]
                   - segmax[(g * NN + d) * NHEADS + h]);
    if (c == 0) atomicAdd(&denom[(g * NN + d) * NHEADS + h], w);
    const float* pl = xl + ((size_t)(b * NN + s) * TT + t) * 256 + h * 64 + c * 8;
    float* pm = msg + (((size_t)g * NN + d) * NHEADS + h) * 64 + c * 8;
    #pragma unroll
    for (int k = 0; k < 8; ++k) atomicAdd(&pm[k], w * pl[k]);
}

// ---------------- GATv2 finalize (+bias, head mean) + LayerNorm -> xs[B,N,T,64] ----------------
__global__ void gat_finalize_ln(const float* __restrict__ msg, const float* __restrict__ denom,
                                const float* __restrict__ gat_bias,
                                const float* __restrict__ g_, const float* __restrict__ b_,
                                float* __restrict__ xs)
{
    int gn = blockIdx.x;                 // g*NN + n
    int d  = threadIdx.x;
    int g  = gn / NN, n = gn % NN;
    int b  = g / TT, t = g % TT;
    float acc = 0.0f;
    #pragma unroll
    for (int h = 0; h < NHEADS; ++h) {
        float den = denom[gn * NHEADS + h] + 1e-16f;
        acc += msg[((size_t)gn * NHEADS + h) * 64 + d] / den;
    }
    float v = acc * (1.0f / NHEADS) + gat_bias[d];
    __shared__ float red[64];
    xs[((size_t)(b * NN + n) * TT + t) * 64 + d] = ln_norm64(v, g_, b_, d, red);
}

// ---------------- temporal MHA core: per (b,n), heads=4, d=16, T=24 ----------------
__global__ void temporal_attn(const float* __restrict__ qkv, float* __restrict__ out)
{
    int bn = blockIdx.x;
    __shared__ float sh[TT * 192];
    const float* base = qkv + (size_t)bn * TT * 192;
    for (int i = threadIdx.x; i < TT * 192; i += blockDim.x) sh[i] = base[i];
    __syncthreads();
    int tq = threadIdx.x % TT;
    int h  = threadIdx.x / TT;
    const float* q = sh + tq * 192 + h * 16;
    float s[TT]; float mx = -1e30f;
    for (int tk = 0; tk < TT; ++tk) {
        const float* kk = sh + tk * 192 + 64 + h * 16;
        float a = 0.0f;
        #pragma unroll
        for (int d = 0; d < 16; ++d) a += q[d] * kk[d];
        s[tk] = a * 0.25f;                  // 1/sqrt(16)
        mx = fmaxf(mx, s[tk]);
    }
    float se = 0.0f;
    for (int tk = 0; tk < TT; ++tk) { s[tk] = expf(s[tk] - mx); se += s[tk]; }
    float inv = 1.0f / se;
    float o[16] = {};
    for (int tk = 0; tk < TT; ++tk) {
        const float* vv = sh + tk * 192 + 128 + h * 16;
        float w = s[tk] * inv;
        #pragma unroll
        for (int d = 0; d < 16; ++d) o[d] += w * vv[d];
    }
    float* op = out + ((size_t)bn * TT + tq) * 64 + h * 16;
    #pragma unroll
    for (int d = 0; d < 16; ++d) op[d] = o[d];
}

// ---------------- residual add + LayerNorm ----------------
__global__ void add_ln(const float* a, const float* b, const float* g_, const float* b_,
                       float* out)
{
    int r = blockIdx.x, d = threadIdx.x;
    size_t idx = (size_t)r * 64 + d;
    float v = a[idx] + b[idx];
    __shared__ float red[64];
    out[idx] = ln_norm64(v, g_, b_, d, red);
}

// ---------------- gate fuse + LayerNorm: x' = LN(sig(g)*xs + (1-sig(g))*xtm + res) ----------------
__global__ void gate_fuse_ln(const float* gbuf, const float* xs, const float* xtm,
                             const float* res, const float* g_, const float* b_,
                             float* out)
{
    int r = blockIdx.x, d = threadIdx.x;
    size_t idx = (size_t)r * 64 + d;
    float a = 1.0f / (1.0f + expf(-gbuf[idx]));
    float v = a * xs[idx] + (1.0f - a) * xtm[idx] + res[idx];
    __shared__ float red[64];
    out[idx] = ln_norm64(v, g_, b_, d, red);
}

// ---------------- mean over T ----------------
__global__ void mean_t(const float* x, float* xm)
{
    int bn = blockIdx.x, d = threadIdx.x;
    float s = 0.0f;
    for (int t = 0; t < TT; ++t) s += x[((size_t)bn * TT + t) * 64 + d];
    xm[(size_t)bn * 64 + d] = s * (1.0f / TT);
}

// ---------------- weather cross attention: q[B,N,64] vs k,v[B,24,64] ----------------
__global__ void cross_attn(const float* __restrict__ q, const float* __restrict__ k,
                           const float* __restrict__ v, float* __restrict__ out)
{
    int bn = blockIdx.x;           // b*NN + n
    int b  = bn / NN;
    __shared__ float shk[TWW * 64];
    __shared__ float shv[TWW * 64];
    for (int i = threadIdx.x; i < TWW * 64; i += blockDim.x) {
        shk[i] = k[(size_t)b * TWW * 64 + i];
        shv[i] = v[(size_t)b * TWW * 64 + i];
    }
    __syncthreads();
    int h = threadIdx.x >> 4;
    int d = threadIdx.x & 15;
    const float* qr = q + (size_t)bn * 64 + h * 16;
    float s[TWW]; float mx = -1e30f;
    for (int tk = 0; tk < TWW; ++tk) {
        const float* kk = shk + tk * 64 + h * 16;
        float a = 0.0f;
        #pragma unroll
        for (int dd = 0; dd < 16; ++dd) a += qr[dd] * kk[dd];
        s[tk] = a * 0.25f;
        mx = fmaxf(mx, s[tk]);
    }
    float se = 0.0f;
    for (int tk = 0; tk < TWW; ++tk) { s[tk] = expf(s[tk] - mx); se += s[tk]; }
    float inv = 1.0f / se;
    float o = 0.0f;
    for (int tk = 0; tk < TWW; ++tk) o += s[tk] * inv * shv[tk * 64 + h * 16 + d];
    out[(size_t)bn * 64 + h * 16 + d] = o;
}

// ================= host launch =================
extern "C" void kernel_launch(void* const* d_in, const int* in_sizes, int n_in,
                              void* d_out, int out_size, void* d_ws, size_t ws_size,
                              hipStream_t stream)
{
    (void)in_sizes; (void)n_in; (void)out_size; (void)ws_size;
    const float* x_traffic = (const float*)d_in[0];
    const float* x_weather = (const float*)d_in[1];
    const float* hour      = (const float*)d_in[2];
    const float* enc_W     = (const float*)d_in[3];
    const float* enc_b     = (const float*)d_in[4];
    const float* dow_tab   = (const float*)d_in[5];
    const float* we_tab    = (const float*)d_in[6];
    const float* tproj_W   = (const float*)d_in[7];
    const float* tproj_b   = (const float*)d_in[8];
    const float* gat_Wl    = (const float*)d_in[9];
    const float* gat_bl    = (const float*)d_in[10];
    const float* gat_Wr    = (const float*)d_in[11];
    const float* gat_br    = (const float*)d_in[12];
    const float* gat_att   = (const float*)d_in[13];
    const float* gat_bias  = (const float*)d_in[14];
    const float* ta_in_W   = (const float*)d_in[15];
    const float* ta_in_b   = (const float*)d_in[16];
    const float* ta_out_W  = (const float*)d_in[17];
    const float* ta_out_b  = (const float*)d_in[18];
    const float* ffn_W1    = (const float*)d_in[19];
    const float* ffn_b1    = (const float*)d_in[20];
    const float* ffn_W2    = (const float*)d_in[21];
    const float* ffn_b2    = (const float*)d_in[22];
    const float* gate_W    = (const float*)d_in[23];
    const float* gate_b    = (const float*)d_in[24];
    const float* ln_sp     = (const float*)d_in[25];
    const float* ln_t1     = (const float*)d_in[26];
    const float* ln_t2     = (const float*)d_in[27];
    const float* ln_f      = (const float*)d_in[28];
    const float* wproj_W   = (const float*)d_in[29];
    const float* wproj_b   = (const float*)d_in[30];
    const float* wa_in_W   = (const float*)d_in[31];
    const float* wa_in_b   = (const float*)d_in[32];
    const float* wa_out_W  = (const float*)d_in[33];
    const float* wa_out_b  = (const float*)d_in[34];
    const float* wnorm_g   = (const float*)d_in[35];
    const float* wnorm_b   = (const float*)d_in[36];
    const float* mu_W      = (const float*)d_in[37];
    const float* mu_b      = (const float*)d_in[38];
    const float* sg_W      = (const float*)d_in[39];
    const float* sg_b      = (const float*)d_in[40];
    const float* pi_W      = (const float*)d_in[41];
    const float* pi_b      = (const float*)d_in[42];
    const int*   edge_idx  = (const int*)d_in[43];
    const int*   dow       = (const int*)d_in[44];
    const int*   is_we     = (const int*)d_in[45];
    float* out = (float*)d_out;

    const int* src = edge_idx;
    const int* dst = edge_idx + NEDGE;

    // ---- workspace arena ----
    float* w = (float*)d_ws;
    size_t off = 0;
    auto take = [&](size_t nf) -> float* {
        float* p = w + off;
        off += (nf + 1023) & ~(size_t)1023;
        return p;
    };
    float* temp   = take((size_t)NGRAPH * HH);
    float* x0     = take((size_t)NROWS * HH);
    float* x1     = take((size_t)NROWS * HH);
    float* bufL   = take((size_t)NROWS * 256);   // xl, later h1
    float* bufR   = take((size_t)NROWS * 256);   // xr, later msg, later qkv
    float* logits = take((size_t)NGRAPH * NEDGE * NHEADS);
    float* segmax = take((size_t)NGRAPH * NN * NHEADS);
    float* denom  = take((size_t)NGRAPH * NN * NHEADS);
    float* xsb    = take((size_t)NROWS * HH);
    float* s1     = take((size_t)NROWS * HH);    // xattn / gatebuf / xm
    float* s2     = take((size_t)NROWS * HH);    // attnout / ffn / wp / xfin
    float* s3     = take((size_t)NROWS * HH);    // xn1 / q
    float* s4     = take((size_t)NROWS * HH);    // xtm / attn proj out
    float* kbuf   = take((size_t)BB * TWW * HH);
    float* vbuf   = take((size_t)BB * TWW * HH);
    // f16 weight copies (2 halves per float slot)
    auto takeh = [&](size_t nh) -> _Float16* {
        return (_Float16*)take((nh + 1) / 2);
    };
    _Float16* hWl   = takeh((size_t)NBLK * 256 * 64);
    _Float16* hWr   = takeh((size_t)NBLK * 256 * 64);
    _Float16* hWqkv = takeh((size_t)NBLK * 192 * 64);
    _Float16* hWo   = takeh((size_t)NBLK * 64 * 64);
    _Float16* hW1   = takeh((size_t)NBLK * 256 * 64);
    _Float16* hW2   = takeh((size_t)NBLK * 64 * 256);
    _Float16* hWg   = takeh((size_t)NBLK * 64 * 128);
    _Float16* hWain = takeh((size_t)192 * 64);
    _Float16* hWaout= takeh((size_t)64 * 64);

    auto cvt = [&](const float* s, _Float16* d, int n) {
        f32_to_f16<<<(n + 255) / 256, 256, 0, stream>>>(s, d, n);
    };
    auto linW = [&](const float* X, const _Float16* Wt, const float* bi, float* O,
                    int M_, int N_, int K_, int ldw_, int accum_, int act_) {
        int waves  = (M_ / 16) * (N_ / 64);      // 4 N-tiles per wave
        int blocks = (waves + 3) / 4;
        wmma_linear<<<blocks, 128, 0, stream>>>(X, Wt, bi, O, M_, N_, K_, ldw_, accum_, act_);
    };
    auto fill = [&](float* p, float v, size_t n) {
        fillf<<<(int)((n + 255) / 256), 256, 0, stream>>>(p, v, n);
    };

    // ---- one-shot weight conversion ----
    cvt(gat_Wl,   hWl,   NBLK * 256 * 64);
    cvt(gat_Wr,   hWr,   NBLK * 256 * 64);
    cvt(ta_in_W,  hWqkv, NBLK * 192 * 64);
    cvt(ta_out_W, hWo,   NBLK * 64 * 64);
    cvt(ffn_W1,   hW1,   NBLK * 256 * 64);
    cvt(ffn_W2,   hW2,   NBLK * 64 * 256);
    cvt(gate_W,   hWg,   NBLK * 64 * 128);
    cvt(wa_in_W,  hWain, 192 * 64);
    cvt(wa_out_W, hWaout, 64 * 64);

    // ---- embedding ----
    temp_embed<<<NGRAPH, 64, 0, stream>>>(hour, dow, is_we, dow_tab, we_tab,
                                          tproj_W, tproj_b, temp);
    encode<<<NROWS, 64, 0, stream>>>(x_traffic, enc_W, enc_b, temp, x0);

    float* xcur = x0;
    float* xnxt = x1;

    for (int i = 0; i < NBLK; ++i) {
        const _Float16* Wl  = hWl   + (size_t)i * 256 * 64;
        const float*    bl  = gat_bl+ (size_t)i * 256;
        const _Float16* Wr  = hWr   + (size_t)i * 256 * 64;
        const float*    br  = gat_br+ (size_t)i * 256;
        const float*    att = gat_att + (size_t)i * NHEADS * 64;
        const float*    gbi = gat_bias+ (size_t)i * 64;
        const _Float16* Wqkv= hWqkv + (size_t)i * 192 * 64;
        const float*    bqkv= ta_in_b + (size_t)i * 192;
        const _Float16* Wo  = hWo   + (size_t)i * 64 * 64;
        const float*    bo  = ta_out_b+ (size_t)i * 64;
        const _Float16* W1  = hW1   + (size_t)i * 256 * 64;
        const float*    b1  = ffn_b1+ (size_t)i * 256;
        const _Float16* W2  = hW2   + (size_t)i * 64 * 256;
        const float*    b2  = ffn_b2+ (size_t)i * 64;
        const _Float16* Wg  = hWg   + (size_t)i * 64 * 128;
        const float*    bg  = gate_b+ (size_t)i * 64;
        const float*    lsp = ln_sp + (size_t)i * 128;
        const float*    lt1 = ln_t1 + (size_t)i * 128;
        const float*    lt2 = ln_t2 + (size_t)i * 128;
        const float*    lf  = ln_f  + (size_t)i * 128;

        // --- GATv2 spatial branch ---
        linW(xcur, Wl, bl, bufL, NROWS, 256, 64, 64, 0, 0);          // xl
        linW(xcur, Wr, br, bufR, NROWS, 256, 64, 64, 0, 0);          // xr
        fill(segmax, -1e30f, (size_t)NGRAPH * NN * NHEADS);
        fill(denom, 0.0f, (size_t)NGRAPH * NN * NHEADS);
        {
            long long nth = (long long)NGRAPH * NEDGE * NHEADS;
            gat_edge_logits<<<(int)((nth + 255) / 256), 256, 0, stream>>>(
                bufL, bufR, att, src, dst, logits, segmax);
        }
        fill(bufR, 0.0f, (size_t)NROWS * 256);                        // msg (reuses xr)
        {
            long long nth = (long long)NGRAPH * NEDGE * NHEADS * 8;
            gat_edge_accum<<<(int)((nth + 255) / 256), 256, 0, stream>>>(
                bufL, src, dst, logits, segmax, denom, bufR);
        }
        gat_finalize_ln<<<NGRAPH * NN, 64, 0, stream>>>(bufR, denom, gbi,
                                                        lsp, lsp + 64, xsb);

        // --- temporal branch ---
        linW(xcur, Wqkv, bqkv, bufR, NROWS, 192, 64, 64, 0, 0);       // qkv
        temporal_attn<<<BB * NN, 96, 0, stream>>>(bufR, s1);          // xattn
        linW(s1, Wo, bo, s2, NROWS, 64, 64, 64, 0, 0);                // attn out
        add_ln<<<NROWS, 64, 0, stream>>>(xcur, s2, lt1, lt1 + 64, s3);// xn1
        linW(s3, W1, b1, bufL, NROWS, 256, 64, 64, 0, 1);             // h1 = gelu
        linW(bufL, W2, b2, s2, NROWS, 64, 256, 256, 0, 0);            // ffn
        add_ln<<<NROWS, 64, 0, stream>>>(s3, s2, lt2, lt2 + 64, s4);  // xtm

        // --- gated fusion: gate = [xs, xtm] @ Wg^T + bg (split, ldw=128) ---
        linW(xsb, Wg,      bg,      s1, NROWS, 64, 64, 128, 0, 0);
        linW(s4,  Wg + 64, nullptr, s1, NROWS, 64, 64, 128, 1, 0);
        gate_fuse_ln<<<NROWS, 64, 0, stream>>>(s1, xsb, s4, xcur, lf, lf + 64, xnxt);

        float* tmpp = xcur; xcur = xnxt; xnxt = tmpp;
    }

    // ---- head: mean over T, weather cross-attention, MDN heads ----
    mean_t<<<BB * NN, 64, 0, stream>>>(xcur, s1);                          // xm
    small_linear<<<(NGRAPH * 64 + 255) / 256, 256, 0, stream>>>(
        x_weather, wproj_W, wproj_b, s2, BB * TWW, 64, 3, 0);              // wp
    linW(s1, hWain,            wa_in_b,       s3,   BB * NN,  64, 64, 64, 0, 0); // q
    linW(s2, hWain + 64 * 64,  wa_in_b + 64,  kbuf, BB * TWW, 64, 64, 64, 0, 0); // k
    linW(s2, hWain + 128 * 64, wa_in_b + 128, vbuf, BB * TWW, 64, 64, 64, 0, 0); // v
    cross_attn<<<BB * NN, 64, 0, stream>>>(s3, kbuf, vbuf, bufR);
    linW(bufR, hWaout, wa_out_b, s4, BB * NN, 64, 64, 64, 0, 0);
    add_ln<<<BB * NN, 64, 0, stream>>>(s1, s4, wnorm_g, wnorm_b, s2);      // xfin

    int hth = (BB * NN * OUTC + 255) / 256;
    small_linear<<<hth, 256, 0, stream>>>(s2, mu_W, mu_b, out,                 BB * NN, OUTC, 64, 0);
    small_linear<<<hth, 256, 0, stream>>>(s2, sg_W, sg_b, out + BB * NN * OUTC, BB * NN, OUTC, 64, 1);
    small_linear<<<hth, 256, 0, stream>>>(s2, pi_W, pi_b, out + 2 * BB * NN * OUTC, BB * NN, OUTC, 64, 0);
}
